// FlowModel_85753317032111
// MI455X (gfx1250) — matrix operand order repaired and checked by reading, not compiled, not run on hardware
//
#include <hip/hip_runtime.h>

// Problem constants (from reference)
#define B_    1024
#define D_    12288
#define H_    6144
#define HID_  512
#define STEPS_ 8

// GEMM tiling
#define BM 128
#define BN 128
#define BK 32
#define LDS_STRIDE 40   // BK + 8 pad (bf16 units); 80B row stride, 16B aligned

typedef __attribute__((ext_vector_type(16))) __bf16 v16bf;
typedef __attribute__((ext_vector_type(8)))  __bf16 v8bf;
typedef __attribute__((ext_vector_type(4)))  __bf16 v4bf;
typedef __attribute__((ext_vector_type(8)))  float  v8f;
typedef __attribute__((ext_vector_type(4)))  float  v4f;

// Load one 128x32 (A) + 32x128 (B) f32 chunk into registers (coalesced v4f).
__device__ __forceinline__ void load_chunk_regs(
    const float* __restrict__ A, int lda,
    const float* __restrict__ Bw, int ldb,
    int tileM, int tileN, int k0, int tid,
    v4f aregs[4], v4f bregs[4])
{
#pragma unroll
    for (int t = 0; t < 4; ++t) {
        const int idx = tid + t * 256;
        const int ar = idx >> 3;            // 8 float4 per 32-wide A row
        const int ac = (idx & 7) << 2;
        aregs[t] = *(const v4f*)(A + (size_t)(tileM + ar) * lda + (k0 + ac));
        const int br = idx >> 5;            // 32 float4 per 128-wide B row
        const int bc = (idx & 31) << 2;
        bregs[t] = *(const v4f*)(Bw + (size_t)(k0 + br) * ldb + (tileN + bc));
    }
}

// Convert f32->bf16 and store: A row-major, B transposed (N-major) in LDS.
__device__ __forceinline__ void store_chunk_lds(
    __bf16 (*__restrict__ As)[LDS_STRIDE],
    __bf16 (*__restrict__ Bt)[LDS_STRIDE],
    int tid, const v4f aregs[4], const v4f bregs[4])
{
#pragma unroll
    for (int t = 0; t < 4; ++t) {
        const int idx = tid + t * 256;
        const int ar = idx >> 3;
        const int ac = (idx & 7) << 2;
        v4bf av;
        av[0] = (__bf16)aregs[t][0];
        av[1] = (__bf16)aregs[t][1];
        av[2] = (__bf16)aregs[t][2];
        av[3] = (__bf16)aregs[t][3];
        *(v4bf*)&As[ar][ac] = av;           // ds_store_b64
        const int br = idx >> 5;
        const int bc = (idx & 31) << 2;
#pragma unroll
        for (int j = 0; j < 4; ++j)
            Bt[bc + j][br] = (__bf16)bregs[t][j];   // transposed scatter
    }
}

// C[M,N] = act(A[M,K] @ B[K,N] + bias[N]); bf16 WMMA, f32 accumulate.
// M = gridDim.y*128, N = gridDim.x*128; K % 32 == 0; all dims multiples of tile.
__global__ __launch_bounds__(256) void gemm_bias_act_wmma(
    const float* __restrict__ A, int lda,
    const float* __restrict__ Bw, int ldb,
    const float* __restrict__ bias,
    float* __restrict__ C, int ldc,
    int K, int do_relu)
{
    __shared__ __bf16 As[2][BM][LDS_STRIDE];
    __shared__ __bf16 Bt[2][BN][LDS_STRIDE];

    const int tid   = threadIdx.x;
    const int wid   = tid >> 5;          // 8 waves (wave32)
    const int lane  = tid & 31;
    const int lhalf = lane & 15;
    const int hi    = lane >> 4;

    const int tileM = blockIdx.y * BM;
    const int tileN = blockIdx.x * BN;
    const int waveM = (wid & 1) * 64;    // 2x4 wave grid, 64x32 per wave
    const int waveN = (wid >> 1) * 32;

    v8f acc[4][2];
#pragma unroll
    for (int mt = 0; mt < 4; ++mt)
#pragma unroll
        for (int nt = 0; nt < 2; ++nt)
            acc[mt][nt] = {};

    const int nk = K / BK;

    // Prologue: stage chunk 0 into buffer 0.
    {
        v4f ar[4], br[4];
        load_chunk_regs(A, lda, Bw, ldb, tileM, tileN, 0, tid, ar, br);
        store_chunk_lds(As[0], Bt[0], tid, ar, br);
    }
    __syncthreads();

    for (int kk = 0; kk < nk; ++kk) {
        const int cb = kk & 1;
        const int nb = cb ^ 1;
        const bool have_next = (kk + 1) < nk;

        // Issue next chunk's global loads before the matrix math.
        v4f aregs[4], bregs[4];
        if (have_next)
            load_chunk_regs(A, lda, Bw, ldb, tileM, tileN, (kk + 1) * BK,
                            tid, aregs, bregs);
        if (kk + 2 < nk) {
            // speculative prefetch of chunk k+2 -> global_prefetch_b8
            __builtin_prefetch(A + (size_t)(tileM + (tid >> 3)) * lda
                                 + (kk + 2) * BK + ((tid & 7) << 2), 0, 1);
            __builtin_prefetch(Bw + (size_t)((kk + 2) * BK + (tid >> 5)) * ldb
                                 + tileN + ((tid & 31) << 2), 0, 1);
        }

        // Build fragments from LDS per the 16-bit WMMA VGPR layouts.
        v16bf afrag[4];
#pragma unroll
        for (int mt = 0; mt < 4; ++mt) {
            const int row = waveM + mt * 16 + lhalf;
            const v8bf lo = *(const v8bf*)&As[cb][row][hi * 8];        // K 0..7 / 8..15
            const v8bf hv = *(const v8bf*)&As[cb][row][16 + hi * 8];   // K 16..23 / 24..31
            afrag[mt] = __builtin_shufflevector(lo, hv,
                0,1,2,3,4,5,6,7,8,9,10,11,12,13,14,15);
        }
        v16bf bfrag[2];
#pragma unroll
        for (int nt = 0; nt < 2; ++nt) {
            const int nrow = waveN + nt * 16 + lhalf;
            const v8bf lo = *(const v8bf*)&Bt[cb][nrow][hi * 8];       // K 0..7 / 8..15
            const v8bf hv = *(const v8bf*)&Bt[cb][nrow][16 + hi * 8];  // K 16..23 / 24..31
            bfrag[nt] = __builtin_shufflevector(lo, hv,
                0,1,2,3,4,5,6,7,8,9,10,11,12,13,14,15);
        }

#pragma unroll
        for (int mt = 0; mt < 4; ++mt)
#pragma unroll
            for (int nt = 0; nt < 2; ++nt)
                acc[mt][nt] = __builtin_amdgcn_wmma_f32_16x16x32_bf16(
                    false, afrag[mt], false, bfrag[nt],
                    (short)0, acc[mt][nt], false, false);

        if (have_next)
            store_chunk_lds(As[nb], Bt[nb], tid, aregs, bregs);
        __syncthreads();
    }

    // Epilogue: C layout — VGPR r: lanes 0-15 -> M=r, lanes 16-31 -> M=r+8; col = lane&15.
#pragma unroll
    for (int nt = 0; nt < 2; ++nt) {
        const int col = tileN + waveN + nt * 16 + lhalf;
        const float bv = bias[col];
#pragma unroll
        for (int mt = 0; mt < 4; ++mt) {
#pragma unroll
            for (int r = 0; r < 8; ++r) {
                const int row = tileM + waveM + mt * 16 + hi * 8 + r;
                float v = acc[mt][nt][r] + bv;
                if (do_relu) v = fmaxf(v, 0.0f);
                C[(size_t)row * ldc + col] = v;
            }
        }
    }
}

// y = concat(x0, x1*exp(log_scale)+shift), then optional column permutation.
// x_next[b, j] = y[b, perm[j]]  (perm == nullptr -> identity)
__global__ __launch_bounds__(256) void coupling_perm_kernel(
    const float* __restrict__ x_cur,
    const float* __restrict__ sls,
    const int* __restrict__ perm,
    float* __restrict__ x_next)
{
    const int idx = blockIdx.x * 256 + threadIdx.x;   // < B_*D_
    const int b  = idx / D_;
    const int jj = idx - b * D_;
    const int src = perm ? perm[jj] : jj;

    const float* xrow = x_cur + (size_t)b * D_;
    const float* srow = sls   + (size_t)b * D_;
    float val;
    if (src < H_) {
        val = xrow[src];                               // identity half
    } else {
        // y1[n] = x1[n]*exp(log_scale[n]) + shift[n], n = src - H
        val = xrow[src] * __expf(srow[src]) + srow[src - H_];
    }
    x_next[(size_t)b * D_ + jj] = val;
}

extern "C" void kernel_launch(void* const* d_in, const int* in_sizes, int n_in,
                              void* d_out, int out_size, void* d_ws, size_t ws_size,
                              hipStream_t stream) {
    (void)in_sizes; (void)n_in; (void)out_size; (void)ws_size;

    const float* x     = (const float*)d_in[0];
    const float* W1    = (const float*)d_in[1];
    const float* b1    = (const float*)d_in[2];
    const float* W2    = (const float*)d_in[3];
    const float* b2    = (const float*)d_in[4];
    const float* W3    = (const float*)d_in[5];
    const float* b3    = (const float*)d_in[6];
    const int*   perms = (const int*)  d_in[7];

    float* ws   = (float*)d_ws;
    float* x_ws = ws;                                  // B*D
    float* sls  = x_ws + (size_t)B_ * D_;              // B*D
    float* h1   = sls  + (size_t)B_ * D_;              // B*HID
    float* h2   = h1   + (size_t)B_ * HID_;            // B*HID
    float* outf = (float*)d_out;

    const dim3 blk(256);
    const dim3 g12(HID_ / BN, B_ / BM);                // 4 x 8
    const dim3 g3 (D_   / BN, B_ / BM);                // 96 x 8
    const int  cblocks = (B_ * D_) / 256;

    const float* cur = x;
    for (int i = 0; i < STEPS_; ++i) {
        const float* Wi1 = W1 + (size_t)i * H_   * HID_;
        const float* Wi2 = W2 + (size_t)i * HID_ * HID_;
        const float* Wi3 = W3 + (size_t)i * HID_ * D_;
        const float* bi1 = b1 + (size_t)i * HID_;
        const float* bi2 = b2 + (size_t)i * HID_;
        const float* bi3 = b3 + (size_t)i * D_;

        // h1 = relu(x0 @ W1 + b1)   (x0 = cur[:, :H], lda = D)
        gemm_bias_act_wmma<<<g12, blk, 0, stream>>>(
            cur, D_, Wi1, HID_, bi1, h1, HID_, H_, 1);
        // h2 = relu(h1 @ W2 + b2)
        gemm_bias_act_wmma<<<g12, blk, 0, stream>>>(
            h1, HID_, Wi2, HID_, bi2, h2, HID_, HID_, 1);
        // sls = h2 @ W3 + b3
        gemm_bias_act_wmma<<<g3, blk, 0, stream>>>(
            h2, HID_, Wi3, D_, bi3, sls, D_, HID_, 0);

        // coupling + permutation; ping-pong so step 8 lands in d_out
        float* nxt = (i & 1) ? outf : x_ws;
        const int* p = (i < STEPS_ - 1) ? (perms + (size_t)i * D_) : nullptr;
        coupling_perm_kernel<<<cblocks, blk, 0, stream>>>(cur, sls, p, nxt);
        cur = nxt;
    }
}
